// PureFieldMLPv2_47098611368129
// MI455X (gfx1250) — compile-verified, matrix operand order checked by reading, not run on hardware
//
#include <hip/hip_runtime.h>
#include <hip/hip_bf16.h>

// ---------------------------------------------------------------------------
// Types for WMMA (gfx1250, wave32)
// ---------------------------------------------------------------------------
typedef __bf16 bf16_t;
typedef __attribute__((ext_vector_type(16))) __bf16 v16bf;
typedef __attribute__((ext_vector_type(8)))  float  v8f;
typedef unsigned int u32x4 __attribute__((ext_vector_type(4)));
typedef int i32x4 __attribute__((ext_vector_type(4)));

struct Frag32 { u32x4 lo, hi; };   // 32 bytes == v16bf

// Problem dims
#define DH 2048
#define DI 8192
#define DR 128
#define DB 2
#define DT 2048
#define DM (DB*DT)      // 4096 rows

// GEMM tiling
#define BM 128
#define BN 64
#define BK 32
#define LDSA 40         // halfs per row (padded, 80B = 16B aligned)
#define LDSB 40

// ---------------------------------------------------------------------------
// CDNA5 async global->LDS copy (ASYNCcnt) with sync fallback
// ---------------------------------------------------------------------------
#if defined(__has_builtin)
#  if __has_builtin(__builtin_amdgcn_global_load_async_to_lds_b128) && \
      __has_builtin(__builtin_amdgcn_s_wait_asynccnt)
#    define HAVE_ASYNC_LDS 1
#  endif
#endif

#ifdef HAVE_ASYNC_LDS
#define GLOBAL_AS __attribute__((address_space(1)))
#define LDS_AS    __attribute__((address_space(3)))
__device__ __forceinline__ void async_cp16(const bf16_t* g, bf16_t* l) {
    __builtin_amdgcn_global_load_async_to_lds_b128(
        (GLOBAL_AS i32x4*)(bf16_t*)g, (LDS_AS i32x4*)l, 0, 0);
}
#endif

struct FetchRegs { u32x4 ra0, ra1, rb0, rb1; };

// Each thread owns: A segs (rows tid/4 and tid/4+64, bytes [(tid&3)*16,+16))
// and one B seg per B matrix (row tid/4, same 16B column window).
template<bool DUAL>
__device__ __forceinline__ void fetch_issue(
    const bf16_t* __restrict__ A, int lda,
    const bf16_t* __restrict__ B0, const bf16_t* __restrict__ B1, int ldb,
    int k, bf16_t* bufA, bf16_t* bufB0, bf16_t* bufB1,
    int ar0, int aseg, FetchRegs& fr)
{
#ifdef HAVE_ASYNC_LDS
    async_cp16(A  + (size_t)ar0        * lda + k + aseg, bufA  + ar0        * LDSA + aseg);
    async_cp16(A  + (size_t)(ar0 + 64) * lda + k + aseg, bufA  + (ar0 + 64) * LDSA + aseg);
    async_cp16(B0 + (size_t)ar0        * ldb + k + aseg, bufB0 + ar0        * LDSB + aseg);
    if (DUAL)
        async_cp16(B1 + (size_t)ar0 * ldb + k + aseg, bufB1 + ar0 * LDSB + aseg);
    (void)fr;
#else
    fr.ra0 = *(const u32x4*)(A  + (size_t)ar0        * lda + k + aseg);
    fr.ra1 = *(const u32x4*)(A  + (size_t)(ar0 + 64) * lda + k + aseg);
    fr.rb0 = *(const u32x4*)(B0 + (size_t)ar0        * ldb + k + aseg);
    if (DUAL)
        fr.rb1 = *(const u32x4*)(B1 + (size_t)ar0 * ldb + k + aseg);
#endif
}

template<bool DUAL>
__device__ __forceinline__ void fetch_commit(
    bf16_t* bufA, bf16_t* bufB0, bf16_t* bufB1,
    int ar0, int aseg, FetchRegs& fr)
{
#ifdef HAVE_ASYNC_LDS
    __builtin_amdgcn_s_wait_asynccnt(0);
    (void)fr; (void)bufA; (void)bufB0; (void)bufB1; (void)ar0; (void)aseg;
#else
    *(u32x4*)(bufA  + ar0        * LDSA + aseg) = fr.ra0;
    *(u32x4*)(bufA  + (ar0 + 64) * LDSA + aseg) = fr.ra1;
    *(u32x4*)(bufB0 + ar0        * LDSB + aseg) = fr.rb0;
    if (DUAL)
        *(u32x4*)(bufB1 + ar0 * LDSB + aseg) = fr.rb1;
#endif
}

// ---------------------------------------------------------------------------
// Fragment load.  16-bit A 16x32 layout (ISA 7.12.2): lane L holds row
// (L&15), K halfs [(L>>4)*8 .. +8) and [16+(L>>4)*8 .. +8).  B identical when
// stored N-major (N x K) -- all weights are pre-transposed.
// ---------------------------------------------------------------------------
__device__ __forceinline__ v16bf load_frag(const bf16_t* p) {
    Frag32 f;
    f.lo = *(const u32x4*)p;
    f.hi = *(const u32x4*)(p + 16);
    return __builtin_bit_cast(v16bf, f);
}

// One k-step: batch all fragment loads (single dscnt wait), then chain WMMAs.
template<bool DUAL>
__device__ __forceinline__ void compute_step(
    const bf16_t* bA, const bf16_t* bB0, const bf16_t* bB1,
    int wave, int frow, int kbase, v8f (&acc0)[4], v8f (&acc1)[4])
{
    v16bf av = load_frag(bA + (wave * 16 + frow) * LDSA + kbase);
    v16bf bv[4];
#pragma unroll
    for (int j = 0; j < 4; ++j)
        bv[j] = load_frag(bB0 + (j * 16 + frow) * LDSB + kbase);
#pragma unroll
    for (int j = 0; j < 4; ++j)
        acc0[j] = __builtin_amdgcn_wmma_f32_16x16x32_bf16(
            false, av, false, bv[j], (short)0, acc0[j], false, false);
    if (DUAL) {
#pragma unroll
        for (int j = 0; j < 4; ++j)
            bv[j] = load_frag(bB1 + (j * 16 + frow) * LDSB + kbase);
#pragma unroll
        for (int j = 0; j < 4; ++j)
            acc1[j] = __builtin_amdgcn_wmma_f32_16x16x32_bf16(
                false, av, false, bv[j], (short)0, acc1[j], false, false);
    }
}

// Double-buffered, software-pipelined k-loop.
// sA/sB0/sB1 are 2x-sized LDS buffers.
template<bool DUAL>
__device__ __forceinline__ void mm_loop(
    const bf16_t* __restrict__ A, int lda,
    const bf16_t* __restrict__ B0, const bf16_t* __restrict__ B1, int ldb,
    int K,
    bf16_t* sA, bf16_t* sB0, bf16_t* sB1,
    v8f (&acc0)[4], v8f (&acc1)[4])
{
    const int tid  = threadIdx.x;
    const int wave = tid >> 5;
    const int lane = tid & 31;
    const int frow  = lane & 15;
    const int kbase = (lane >> 4) * 8;
    const int ar0  = tid >> 2;
    const int aseg = (tid & 3) * 8;

    FetchRegs fr;
    fetch_issue<DUAL>(A, lda, B0, B1, ldb, 0, sA, sB0, sB1, ar0, aseg, fr);
    fetch_commit<DUAL>(sA, sB0, sB1, ar0, aseg, fr);
    __syncthreads();

    int cur = 0;
    for (int k0 = 0; k0 < K - BK; k0 += BK) {
        const int nxt = cur ^ 1;
        bf16_t* nA  = sA  + nxt * BM * LDSA;
        bf16_t* nB0 = sB0 + nxt * BN * LDSB;
        bf16_t* nB1 = sB1 + nxt * BN * LDSB;
        fetch_issue<DUAL>(A, lda, B0, B1, ldb, k0 + BK, nA, nB0, nB1, ar0, aseg, fr);
        compute_step<DUAL>(sA + cur * BM * LDSA, sB0 + cur * BN * LDSB,
                           sB1 + cur * BN * LDSB, wave, frow, kbase, acc0, acc1);
        fetch_commit<DUAL>(nA, nB0, nB1, ar0, aseg, fr);
        __syncthreads();
        cur = nxt;
    }
    compute_step<DUAL>(sA + cur * BM * LDSA, sB0 + cur * BN * LDSB,
                       sB1 + cur * BN * LDSB, wave, frow, kbase, acc0, acc1);
    __syncthreads();
}

// ---------------------------------------------------------------------------
// Generic WMMA GEMM: C(M x N) = A(M x K) * B^T(N x K); f32 or bf16 output.
// grid = (N/BN, M/BM), block = 256 (8 waves; wave w owns rows w*16..+16).
// ---------------------------------------------------------------------------
template<bool BF16OUT>
__global__ __launch_bounds__(256) void k_gemm(
    const bf16_t* __restrict__ A, int lda,
    const bf16_t* __restrict__ Bm, int K,
    void* __restrict__ Cv, int ldc)
{
    __shared__ __attribute__((aligned(16))) bf16_t sA[2 * BM * LDSA];
    __shared__ __attribute__((aligned(16))) bf16_t sB[2 * BN * LDSB];

    const int blockM = blockIdx.y * BM;
    const int blockN = blockIdx.x * BN;
    v8f acc[4] = {};
    v8f dummy[4];

    mm_loop<false>(A + (size_t)blockM * lda, lda,
                   Bm + (size_t)blockN * K, Bm + (size_t)blockN * K, K, K,
                   sA, sB, sB, acc, dummy);

    const int wave = threadIdx.x >> 5, lane = threadIdx.x & 31;
    const int rowbase = blockM + wave * 16 + ((lane >> 4) * 8);
#pragma unroll
    for (int j = 0; j < 4; ++j) {
        int col = blockN + j * 16 + (lane & 15);
#pragma unroll
        for (int r = 0; r < 8; ++r) {
            size_t off = (size_t)(rowbase + r) * ldc + col;
            float v = acc[j][r];
            if (BF16OUT) ((bf16_t*)Cv)[off] = (bf16_t)v;
            else         ((float*)Cv)[off]  = v;
        }
    }
}

// ---------------------------------------------------------------------------
// Fused mid kernel: per (m,n) tile over N = I:
//   gate = x@WgT, up = x@WuT     (K=2048, dual-B shares the A fragment)
//   graw = A1@WgbT, gup = A2@WubT (K=128)
//   Hbuf = silu(gate)*up ; Gbuf = mixture(graw; rw[b]) * gup   (bf16)
// ---------------------------------------------------------------------------
__global__ __launch_bounds__(256) void k_fused_mid(
    const bf16_t* __restrict__ xb,   const bf16_t* __restrict__ wgT,
    const bf16_t* __restrict__ wuT,  const bf16_t* __restrict__ A12,
    const bf16_t* __restrict__ wgbT, const bf16_t* __restrict__ wubT,
    const float*  __restrict__ rw,
    bf16_t* __restrict__ Hbuf, bf16_t* __restrict__ Gbuf)
{
    __shared__ __attribute__((aligned(16))) bf16_t sA [2 * BM * LDSA];
    __shared__ __attribute__((aligned(16))) bf16_t sB0[2 * BN * LDSB];
    __shared__ __attribute__((aligned(16))) bf16_t sB1[2 * BN * LDSB];

    const int blockM = blockIdx.y * BM;
    const int blockN = blockIdx.x * BN;

    v8f acc_g[4] = {}, acc_u[4] = {};
    mm_loop<true>(xb + (size_t)blockM * DH, DH,
                  wgT + (size_t)blockN * DH, wuT + (size_t)blockN * DH, DH, DH,
                  sA, sB0, sB1, acc_g, acc_u);

    v8f acc_gr[4] = {}, acc_gu[4] = {};
    v8f dummy[4];
    mm_loop<false>(A12 + (size_t)blockM * (2 * DR), 2 * DR,
                   wgbT + (size_t)blockN * DR, wgbT + (size_t)blockN * DR, DR, DR,
                   sA, sB0, sB0, acc_gr, dummy);
    mm_loop<false>(A12 + (size_t)blockM * (2 * DR) + DR, 2 * DR,
                   wubT + (size_t)blockN * DR, wubT + (size_t)blockN * DR, DR, DR,
                   sA, sB0, sB0, acc_gu, dummy);

    const int wave = threadIdx.x >> 5, lane = threadIdx.x & 31;
    const int rowbase = blockM + wave * 16 + ((lane >> 4) * 8);
#pragma unroll
    for (int j = 0; j < 4; ++j) {
        int col = blockN + j * 16 + (lane & 15);
#pragma unroll
        for (int r = 0; r < 8; ++r) {
            int row = rowbase + r;
            int b = row >> 11;                 // row / T
            float g  = acc_g[j][r],  u  = acc_u[j][r];
            float gr = acc_gr[j][r], gu = acc_gu[j][r];
            float sig_g = 1.0f / (1.0f + expf(-g));
            float h = g * sig_g * u;           // silu(gate) * up
            float sg = 1.0f / (1.0f + expf(-gr));
            float gg = rw[b * 4 + 0] * fmaxf(gr, 0.0f)
                     + rw[b * 4 + 1] * tanhf(gr)
                     + rw[b * 4 + 2] * (2.0f * sg - 1.0f)
                     + rw[b * 4 + 3] * (gr * sg);
            size_t off = (size_t)row * DI + col;
            Hbuf[off] = (bf16_t)h;
            Gbuf[off] = (bf16_t)(gg * gu);
        }
    }
}

// ---------------------------------------------------------------------------
// Small utility kernels
// ---------------------------------------------------------------------------
__global__ void k_cvt(const float* __restrict__ in, bf16_t* __restrict__ out, size_t n) {
    size_t i = (size_t)blockIdx.x * blockDim.x + threadIdx.x;
    size_t stride = (size_t)gridDim.x * blockDim.x;
    for (; i < n; i += stride) out[i] = (bf16_t)in[i];
}

// Transpose + convert: in f32 (R x C) -> out bf16 (C x R). dims multiple of 32.
__global__ __launch_bounds__(256) void k_transpose_cvt(
    const float* __restrict__ in, bf16_t* __restrict__ out, int R, int C)
{
    __shared__ float tile[32][33];
    const int c0 = blockIdx.x * 32, r0 = blockIdx.y * 32;
    const int tx = threadIdx.x & 31, ty = threadIdx.x >> 5;
    for (int i = ty; i < 32; i += 8)
        tile[i][tx] = in[(size_t)(r0 + i) * C + c0 + tx];
    __syncthreads();
    for (int i = ty; i < 32; i += 8)
        out[(size_t)(c0 + i) * R + r0 + tx] = (bf16_t)tile[tx][i];
}

// rule_input[b,h] = mean over T of x[b,t,h].  grid = (B*H)/256
__global__ void k_rmean(const float* __restrict__ x, float* __restrict__ rmean) {
    int idx = blockIdx.x * 256 + threadIdx.x;         // 0..B*H-1
    int b = idx >> 11, h = idx & (DH - 1);
    const float* p = x + (size_t)b * DT * DH + h;
    float s = 0.f;
    for (int t = 0; t < DT; ++t) s += p[(size_t)t * DH];
    rmean[idx] = s * (1.0f / (float)DT);
}

// rw[b,:] = softmax((rmean[b] @ w_rule) / ln2).  1 block, 8 waves = 8 (b,r).
__global__ void k_rw(const float* __restrict__ rmean,
                     const float* __restrict__ w_rule, float* __restrict__ rw)
{
    __shared__ float dot[8];
    const int tid = threadIdx.x, wave = tid >> 5, lane = tid & 31;
    const int b = wave >> 2, r = wave & 3;
    float s = 0.f;
    for (int h = lane; h < DH; h += 32)
        s += rmean[b * DH + h] * w_rule[h * 4 + r];
#pragma unroll
    for (int o = 16; o > 0; o >>= 1) s += __shfl_xor(s, o, 32);
    if (lane == 0) dot[wave] = s * (1.0f / 0.693147180559945f);
    __syncthreads();
    if (tid < 2) {
        float d0 = dot[tid*4+0], d1 = dot[tid*4+1], d2 = dot[tid*4+2], d3 = dot[tid*4+3];
        float m = fmaxf(fmaxf(d0, d1), fmaxf(d2, d3));
        float e0 = expf(d0 - m), e1 = expf(d1 - m), e2 = expf(d2 - m), e3 = expf(d3 - m);
        float inv = 1.0f / (e0 + e1 + e2 + e3);
        rw[tid*4+0] = e0*inv; rw[tid*4+1] = e1*inv; rw[tid*4+2] = e2*inv; rw[tid*4+3] = e3*inv;
    }
}

// out = orig + alpha*(pf_scale*conv3(pf) - orig), conv over channel dim H.
__global__ void k_final(float* __restrict__ out, const float* __restrict__ pf,
                        const float* __restrict__ alpha_p, const float* __restrict__ pfs_p,
                        const float* __restrict__ ca, size_t total)
{
    const float alpha = *alpha_p, pfs = *pfs_p;
    const float c0 = ca[0], c1 = ca[1], c2 = ca[2];
    size_t i = (size_t)blockIdx.x * blockDim.x + threadIdx.x;
    size_t stride = (size_t)gridDim.x * blockDim.x;
    for (; i < total; i += stride) {
        int h = (int)(i & (DH - 1));
        float left  = (h > 0)       ? pf[i - 1] : 0.f;
        float right = (h < DH - 1)  ? pf[i + 1] : 0.f;
        float v = (c0 * left + c1 * pf[i] + c2 * right) * pfs;
        float o = out[i];
        out[i] = o + alpha * (v - o);
    }
}

// ---------------------------------------------------------------------------
// Host launcher
// ---------------------------------------------------------------------------
extern "C" void kernel_launch(void* const* d_in, const int* in_sizes, int n_in,
                              void* d_out, int out_size, void* d_ws, size_t ws_size,
                              hipStream_t stream) {
    const float* x       = (const float*)d_in[0];
    const float* w_gate  = (const float*)d_in[1];   // (H, I)
    const float* w_up    = (const float*)d_in[2];   // (H, I)
    const float* w_down  = (const float*)d_in[3];   // (I, H)
    const float* w_ga    = (const float*)d_in[4];   // (H, R)
    const float* w_gb    = (const float*)d_in[5];   // (R, I)
    const float* w_ua    = (const float*)d_in[6];   // (H, R)
    const float* w_ub    = (const float*)d_in[7];   // (R, I)
    const float* w_da    = (const float*)d_in[8];   // (I, R)
    const float* w_db    = (const float*)d_in[9];   // (R, H)
    const float* w_rule  = (const float*)d_in[10];  // (H, NR)
    const float* alpha   = (const float*)d_in[11];
    const float* pfscale = (const float*)d_in[12];
    const float* ca      = (const float*)d_in[13];
    float* out = (float*)d_out;
    (void)in_sizes; (void)n_in; (void)out_size; (void)ws_size;

    char* ws = (char*)d_ws;
    auto alloc = [&](size_t bytes) -> char* {
        char* p = ws; ws += (bytes + 255) & ~(size_t)255; return p;
    };
    bf16_t* xb    = (bf16_t*)alloc((size_t)DM * DH * 2);
    bf16_t* wgT   = (bf16_t*)alloc((size_t)DI * DH * 2);
    bf16_t* wuT   = (bf16_t*)alloc((size_t)DI * DH * 2);
    bf16_t* wdT   = (bf16_t*)alloc((size_t)DH * DI * 2);
    bf16_t* wabT  = (bf16_t*)alloc((size_t)2 * DR * DH * 2);
    bf16_t* wgbT  = (bf16_t*)alloc((size_t)DI * DR * 2);
    bf16_t* wubT  = (bf16_t*)alloc((size_t)DI * DR * 2);
    bf16_t* wdaT  = (bf16_t*)alloc((size_t)DR * DI * 2);
    bf16_t* wdbT  = (bf16_t*)alloc((size_t)DH * DR * 2);
    bf16_t* A12   = (bf16_t*)alloc((size_t)DM * 2 * DR * 2);
    bf16_t* Hbuf  = (bf16_t*)alloc((size_t)DM * DI * 2);
    bf16_t* Gbuf  = (bf16_t*)alloc((size_t)DM * DI * 2);
    bf16_t* Pbuf  = (bf16_t*)alloc((size_t)DM * DR * 2);
    float*  PFbuf = (float*) alloc((size_t)DM * DH * 4);
    float*  rmean = (float*) alloc((size_t)DB * DH * 4);
    float*  rw    = (float*) alloc((size_t)DB * 4 * 4);

    // 1) conversions / transposes
    k_cvt<<<8192, 256, 0, stream>>>(x, xb, (size_t)DM * DH);
    k_transpose_cvt<<<dim3(DI/32, DH/32), 256, 0, stream>>>(w_gate, wgT, DH, DI);
    k_transpose_cvt<<<dim3(DI/32, DH/32), 256, 0, stream>>>(w_up,   wuT, DH, DI);
    k_transpose_cvt<<<dim3(DH/32, DI/32), 256, 0, stream>>>(w_down, wdT, DI, DH);
    k_transpose_cvt<<<dim3(DR/32, DH/32), 256, 0, stream>>>(w_ga, wabT,            DH, DR);
    k_transpose_cvt<<<dim3(DR/32, DH/32), 256, 0, stream>>>(w_ua, wabT + (size_t)DR*DH, DH, DR);
    k_transpose_cvt<<<dim3(DI/32, DR/32), 256, 0, stream>>>(w_gb, wgbT, DR, DI);
    k_transpose_cvt<<<dim3(DI/32, DR/32), 256, 0, stream>>>(w_ub, wubT, DR, DI);
    k_transpose_cvt<<<dim3(DR/32, DI/32), 256, 0, stream>>>(w_da, wdaT, DI, DR);
    k_transpose_cvt<<<dim3(DH/32, DR/32), 256, 0, stream>>>(w_db, wdbT, DR, DH);

    // 2) rule weights
    k_rmean<<<(DB * DH) / 256, 256, 0, stream>>>(x, rmean);
    k_rw<<<1, 256, 0, stream>>>(rmean, w_rule, rw);

    // 3) A12 = x @ [w_gate_a | w_up_a]   (M=4096, N=256, K=2048)
    k_gemm<true><<<dim3((2*DR)/BN, DM/BM), 256, 0, stream>>>(
        xb, DH, wabT, DH, A12, 2 * DR);

    // 4) fused mid: gate/up + low-rank graw/gup + activations -> Hbuf, Gbuf
    k_fused_mid<<<dim3(DI/BN, DM/BM), 256, 0, stream>>>(
        xb, wgT, wuT, A12, wgbT, wubT, rw, Hbuf, Gbuf);

    // 5) orig_out = Hbuf @ w_mlp_down  (f32, directly into d_out)
    k_gemm<false><<<dim3(DH/BN, DM/BM), 256, 0, stream>>>(
        Hbuf, DI, wdT, DI, out, DH);

    // 6) P = Gbuf @ w_down_a  (M=4096, N=128, K=8192)
    k_gemm<true><<<dim3(DR/BN, DM/BM), 256, 0, stream>>>(
        Gbuf, DI, wdaT, DI, Pbuf, DR);

    // 7) pf = P @ w_down_b   (M=4096, N=2048, K=128)
    k_gemm<false><<<dim3(DH/BN, DM/BM), 256, 0, stream>>>(
        Pbuf, DR, wdbT, DR, PFbuf, DH);

    // 8) conv3 along H + alpha blend (in place on d_out)
    k_final<<<4096, 256, 0, stream>>>(out, PFbuf, alpha, pfscale, ca,
                                      (size_t)DM * DH);
}
